// DecoderRNN_88338887344425
// MI455X (gfx1250) — matrix operand order repaired and checked by reading, not compiled, not run on hardware
//
#include <hip/hip_runtime.h>
#include <hip/hip_bf16.h>
#include <math.h>

#define B_ 32
#define S_ 2048
#define H_ 1024
#define E_ 1024
#define V_ 50257
#define L_ 4

typedef __attribute__((ext_vector_type(16))) __bf16 v16bf;
typedef __attribute__((ext_vector_type(8)))  __bf16 v8bf;
typedef __attribute__((ext_vector_type(8)))  float  v8f;
typedef __attribute__((ext_vector_type(4)))  float  v4f;

__device__ __forceinline__ float sigm(float x) { return 1.0f / (1.0f + expf(-x)); }

// ---------------------------------------------------------------------------
// Embedding gather: x = [emb[input_seq[b]], emb[keyword[b]]]  (B x 2E)
// ---------------------------------------------------------------------------
__global__ __launch_bounds__(256) void gather_embed(
    const int* __restrict__ input_seq, const int* __restrict__ keyword,
    const float* __restrict__ emb, float* __restrict__ x_f32,
    __bf16* __restrict__ x_bf) {
  int i = blockIdx.x * 256 + threadIdx.x;      // b*2E + col
  int b = i >> 11;
  int col = i & (2 * E_ - 1);
  int tok = (col < E_) ? input_seq[b] : keyword[b];
  int cc = (col < E_) ? col : col - E_;
  float val = emb[(size_t)tok * E_ + cc];
  x_f32[i] = val;
  x_bf[i] = (__bf16)val;
}

__global__ __launch_bounds__(256) void cvt_bf16(
    const float* __restrict__ src, __bf16* __restrict__ dst, int n) {
  int i = blockIdx.x * 256 + threadIdx.x;
  if (i < n) dst[i] = (__bf16)src[i];
}

// ---------------------------------------------------------------------------
// Generic skinny GEMM for M = 32 using v_wmma_f32_16x16x32_bf16.
//   C[32 x Nvalid] = act( A1(32xK1)@W1^T + A2(32xK2)@W2^T + bias0 + bias1 )
// W row-major (N x K, fp32): streamed with non-temporal b128 loads and
// converted to bf16 in registers. Out-of-range N columns are handled by
// CLAMPING the load row (branch-free inner loop; EXEC stays full for WMMA)
// and guarding only the store — an invalid lane's B column only feeds that
// lane's C column, which is never written.
// A/B fragments use contiguous k = 16*half + s in both operands (wave32).
// ---------------------------------------------------------------------------
__global__ __launch_bounds__(256) void gemm32_dual_bf16(
    const __bf16* __restrict__ A1, const float* __restrict__ W1, int K1,
    const __bf16* __restrict__ A2, const float* __restrict__ W2, int K2,
    const float* __restrict__ bias0, const float* __restrict__ bias1,
    int Npad, int Nvalid, int ldc, float* __restrict__ Cout, int act) {
  const int lane = threadIdx.x & 31;
  const int wave = threadIdx.x >> 5;
  const int tileN = blockIdx.x * 8 + wave;
  const int n0 = tileN * 16;
  if (n0 >= Npad) return;
  const int nc = n0 + (lane & 15);          // column for B/C fragments
  const int nld = nc < Nvalid ? nc : (Nvalid - 1);  // clamped load row
  const int half = lane >> 4;               // K-half selector
  const int mLo = lane & 15;                // A row (low M tile)

  v8f accLo = {0.f, 0.f, 0.f, 0.f, 0.f, 0.f, 0.f, 0.f};
  v8f accHi = {0.f, 0.f, 0.f, 0.f, 0.f, 0.f, 0.f, 0.f};

  for (int pass = 0; pass < 2; ++pass) {
    const __bf16* A = pass ? A2 : A1;
    const float* W = pass ? W2 : W1;
    const int K = pass ? K2 : K1;
    if (A == nullptr) continue;
    const float* wrow = W + (size_t)nld * K + half * 16;
    const __bf16* arow0 = A + (size_t)mLo * K + half * 16;
    const __bf16* arow1 = A + (size_t)(mLo + 16) * K + half * 16;
    for (int kb = 0; kb < K; kb += 32) {
      // ---- B fragment: lane holds W[nld, kb + half*16 .. +15] (NT stream) ----
      const v4f* wp = (const v4f*)(wrow + kb);
      v4f w0 = __builtin_nontemporal_load(wp + 0);
      v4f w1 = __builtin_nontemporal_load(wp + 1);
      v4f w2 = __builtin_nontemporal_load(wp + 2);
      v4f w3 = __builtin_nontemporal_load(wp + 3);
      v16bf bf;
#pragma unroll
      for (int t = 0; t < 4; ++t) {
        bf[t] = (__bf16)w0[t];
        bf[4 + t] = (__bf16)w1[t];
        bf[8 + t] = (__bf16)w2[t];
        bf[12 + t] = (__bf16)w3[t];
      }
      // ---- A fragment, low M tile: rows 0..15 ----
      {
        v8bf aA = *(const v8bf*)(arow0 + kb);
        v8bf aB = *(const v8bf*)(arow0 + kb + 8);
        v16bf af;
#pragma unroll
        for (int t = 0; t < 8; ++t) { af[t] = aA[t]; af[8 + t] = aB[t]; }
        accLo = __builtin_amdgcn_wmma_f32_16x16x32_bf16(
            false, af, false, bf, (short)0, accLo, false, false);
      }
      // ---- A fragment, high M tile: rows 16..31 ----
      {
        v8bf aA = *(const v8bf*)(arow1 + kb);
        v8bf aB = *(const v8bf*)(arow1 + kb + 8);
        v16bf af;
#pragma unroll
        for (int t = 0; t < 8; ++t) { af[t] = aA[t]; af[8 + t] = aB[t]; }
        accHi = __builtin_amdgcn_wmma_f32_16x16x32_bf16(
            false, af, false, bf, (short)0, accHi, false, false);
      }
    }
  }

  if (nc < Nvalid) {
    float bias = (bias0 ? bias0[nc] : 0.f) + (bias1 ? bias1[nc] : 0.f);
#pragma unroll
    for (int r = 0; r < 8; ++r) {
      int m = r + half * 8;  // C layout: VGPR r, lanes 16-31 offset by 8 rows
      float vLo = accLo[r] + bias;
      float vHi = accHi[r] + bias;
      if (act == 1) { vLo = tanhf(vLo); vHi = tanhf(vHi); }
      Cout[(size_t)m * ldc + nc] = vLo;
      Cout[(size_t)(m + 16) * ldc + nc] = vHi;
    }
  }
}

// ---------------------------------------------------------------------------
// LSTM pointwise: gates (B x 4H, order i,f,g,o) + c -> h2, c2
// ---------------------------------------------------------------------------
__global__ __launch_bounds__(256) void lstm_pointwise(
    const float* __restrict__ gates, const float* __restrict__ c_in,
    float* __restrict__ hn, float* __restrict__ cn,
    __bf16* __restrict__ h_bf, float* __restrict__ h_f32) {
  int i = blockIdx.x * 256 + threadIdx.x;  // b*H + j
  int b = i >> 10;
  int j = i & (H_ - 1);
  const float* g = gates + (size_t)b * 4 * H_;
  float gi = g[j], gf = g[H_ + j], gg = g[2 * H_ + j], go = g[3 * H_ + j];
  float c2 = sigm(gf) * c_in[i] + sigm(gi) * tanhf(gg);
  float h2 = sigm(go) * tanhf(c2);
  cn[i] = c2;
  hn[i] = h2;
  h_bf[i] = (__bf16)h2;
  h_f32[i] = h2;
}

// ---------------------------------------------------------------------------
// Attention prep: uh[k] = sum_j v[j]*Wa[j][k] (k<H) ; ue[k-H] for k>=H.
// ---------------------------------------------------------------------------
__global__ __launch_bounds__(256) void attn_prep(
    const float* __restrict__ Wa, const float* __restrict__ v,
    float* __restrict__ uh, float* __restrict__ ue) {
  int k = blockIdx.x * 256 + threadIdx.x;  // 0..2047
  float acc = 0.f;
  for (int j = 0; j < H_; ++j) acc += v[j] * Wa[(size_t)j * (2 * H_) + k];
  if (k < H_) uh[k] = acc; else ue[k - H_] = acc;
}

// qconst[b] = hlast[b]·uh + ba·v
__global__ __launch_bounds__(256) void qconst_kernel(
    const float* __restrict__ hlast, const float* __restrict__ uh,
    const float* __restrict__ ba, const float* __restrict__ v,
    float* __restrict__ qc) {
  int lane = threadIdx.x & 31;
  int b = blockIdx.x * 8 + (threadIdx.x >> 5);
  float acc = 0.f;
  for (int k = lane; k < H_; k += 32)
    acc += hlast[(size_t)b * H_ + k] * uh[k] + ba[k] * v[k];
  for (int off = 16; off > 0; off >>= 1) acc += __shfl_xor(acc, off, 32);
  if (lane == 0) qc[b] = acc;
}

// energy[b,s] = enc[b,s,:]·ue + qconst[b]   (one wave per row, streams 268MB)
__global__ __launch_bounds__(256) void energy_kernel(
    const float* __restrict__ enc, const float* __restrict__ ue,
    const float* __restrict__ qc, float* __restrict__ energy) {
  __shared__ float su[H_];
  for (int i = threadIdx.x; i < H_; i += 256) su[i] = ue[i];
  __syncthreads();
  int row = blockIdx.x * 8 + (threadIdx.x >> 5);  // b*S + s
  int lane = threadIdx.x & 31;
  const float* e = enc + (size_t)row * H_;
  float acc = 0.f;
  for (int j = 0; j < H_; j += 128) {
    v4f x = *(const v4f*)(e + j + lane * 4);
    int k = j + lane * 4;
    acc += x[0] * su[k] + x[1] * su[k + 1] + x[2] * su[k + 2] + x[3] * su[k + 3];
  }
  for (int off = 16; off > 0; off >>= 1) acc += __shfl_xor(acc, off, 32);
  if (lane == 0) energy[row] = acc + qc[row >> 11];
}

__global__ __launch_bounds__(256) void softmax_kernel(
    const float* __restrict__ energy, float* __restrict__ attn) {
  __shared__ float red[256];
  __shared__ float stat[2];
  int b = blockIdx.x;
  const float* e = energy + (size_t)b * S_;
  float m = -1e30f;
  for (int s = threadIdx.x; s < S_; s += 256) m = fmaxf(m, e[s]);
  red[threadIdx.x] = m;
  __syncthreads();
  for (int st = 128; st > 0; st >>= 1) {
    if (threadIdx.x < st) red[threadIdx.x] = fmaxf(red[threadIdx.x], red[threadIdx.x + st]);
    __syncthreads();
  }
  if (threadIdx.x == 0) stat[0] = red[0];
  __syncthreads();
  float mv = stat[0];
  float sum = 0.f;
  for (int s = threadIdx.x; s < S_; s += 256) sum += expf(e[s] - mv);
  red[threadIdx.x] = sum;
  __syncthreads();
  for (int st = 128; st > 0; st >>= 1) {
    if (threadIdx.x < st) red[threadIdx.x] += red[threadIdx.x + st];
    __syncthreads();
  }
  if (threadIdx.x == 0) stat[1] = red[0];
  __syncthreads();
  float inv = 1.f / stat[1];
  for (int s = threadIdx.x; s < S_; s += 256)
    attn[(size_t)b * S_ + s] = expf(e[s] - mv) * inv;
}

// context partials over S-chunks (deterministic two-stage reduction)
__global__ __launch_bounds__(256) void context_partial(
    const float* __restrict__ enc, const float* __restrict__ attn,
    float* __restrict__ part) {
  int chunk = blockIdx.x, b = blockIdx.y;
  int h4 = threadIdx.x * 4;
  const float* eb = enc + (size_t)b * S_ * H_;
  v4f acc = {0.f, 0.f, 0.f, 0.f};
  int s0 = chunk * (S_ / 16);
  for (int s = s0; s < s0 + S_ / 16; ++s) {
    float a = attn[(size_t)b * S_ + s];
    v4f x = *(const v4f*)(eb + (size_t)s * H_ + h4);
    __builtin_prefetch(eb + (size_t)(s + 4) * H_ + h4, 0, 1);
    acc += a * x;
  }
  *(v4f*)(part + ((size_t)(chunk * B_ + b)) * H_ + h4) = acc;
}

__global__ __launch_bounds__(256) void context_reduce(
    const float* __restrict__ part, float* __restrict__ ctx) {
  int i = blockIdx.x * 256 + threadIdx.x;  // b*H + h
  int b = i >> 10, h = i & (H_ - 1);
  float s = 0.f;
  for (int c = 0; c < 16; ++c) s += part[((size_t)(c * B_ + b)) * H_ + h];
  ctx[i] = s;
}

// tmp = [h_last | context | key_e]  (B x 3072), bf16 for the Wal GEMM
__global__ __launch_bounds__(256) void build_tmp(
    const float* __restrict__ hlast, const float* __restrict__ ctx,
    const float* __restrict__ x_f32, __bf16* __restrict__ tmp_bf) {
  int i = blockIdx.x * 256 + threadIdx.x;  // b*3072 + c
  int b = i / 3072, c = i % 3072;
  float val;
  if (c < H_) val = hlast[(size_t)b * H_ + c];
  else if (c < 2 * H_) val = ctx[(size_t)b * H_ + (c - H_)];
  else val = x_f32[(size_t)b * 2 * E_ + E_ + (c - 2 * H_)];
  tmp_bf[i] = (__bf16)val;
}

// ---------------------------------------------------------------------------
extern "C" void kernel_launch(void* const* d_in, const int* in_sizes, int n_in,
                              void* d_out, int out_size, void* d_ws, size_t ws_size,
                              hipStream_t stream) {
  const float* enc = (const float*)d_in[0];
  const int* input_seq = (const int*)d_in[1];
  const int* keyword = (const int*)d_in[2];
  const float* h0 = (const float*)d_in[3];
  const float* c0 = (const float*)d_in[4];
  const float* emb = (const float*)d_in[5];
  const float* Wih0 = (const float*)d_in[6];
  const float* Whh0 = (const float*)d_in[7];
  const float* bih0 = (const float*)d_in[8];
  const float* bhh0 = (const float*)d_in[9];
  const float* Wih_r = (const float*)d_in[10];
  const float* Whh_r = (const float*)d_in[11];
  const float* bih_r = (const float*)d_in[12];
  const float* bhh_r = (const float*)d_in[13];
  const float* Wa = (const float*)d_in[14];
  const float* ba = (const float*)d_in[15];
  const float* vv = (const float*)d_in[16];
  const float* Wal = (const float*)d_in[17];
  const float* bal = (const float*)d_in[18];
  const float* Wout = (const float*)d_in[19];
  const float* bout = (const float*)d_in[20];

  float* out = (float*)d_out;
  float* logits = out;                                   // B*V
  float* hn = logits + (size_t)B_ * V_;                  // L*B*H
  float* cn = hn + (size_t)L_ * B_ * H_;                 // L*B*H
  float* attns = cn + (size_t)L_ * B_ * H_;              // B*S

  // workspace carve-out (256B aligned)
  size_t off = 0;
  char* base = (char*)d_ws;
  auto carve = [&](size_t bytes) -> void* {
    void* p = base + off;
    off += (bytes + 255) & ~(size_t)255;
    return p;
  };
  float* x_f32 = (float*)carve((size_t)B_ * 2 * E_ * 4);
  __bf16* x_bf = (__bf16*)carve((size_t)B_ * 2 * E_ * 2);
  __bf16* h0_bf = (__bf16*)carve((size_t)L_ * B_ * H_ * 2);
  float* gates = (float*)carve((size_t)B_ * 4 * H_ * 4);
  __bf16* h_bf = (__bf16*)carve((size_t)B_ * H_ * 2);
  float* h_f32 = (float*)carve((size_t)B_ * H_ * 4);
  float* uh = (float*)carve(H_ * 4);
  float* ue = (float*)carve(H_ * 4);
  float* qc = (float*)carve(B_ * 4);
  float* energy = (float*)carve((size_t)B_ * S_ * 4);
  float* part = (float*)carve((size_t)16 * B_ * H_ * 4);
  float* ctx = (float*)carve((size_t)B_ * H_ * 4);
  __bf16* tmp_bf = (__bf16*)carve((size_t)B_ * 3072 * 2);
  float* hidden_f32 = (float*)carve((size_t)B_ * H_ * 4);
  __bf16* hidden_bf = (__bf16*)carve((size_t)B_ * H_ * 2);
  (void)ws_size; (void)in_sizes; (void)n_in; (void)out_size;

  // 1. embeddings
  gather_embed<<<(B_ * 2 * E_) / 256, 256, 0, stream>>>(input_seq, keyword, emb, x_f32, x_bf);
  // 2. h0 -> bf16 (all layers)
  cvt_bf16<<<(L_ * B_ * H_) / 256, 256, 0, stream>>>(h0, h0_bf, L_ * B_ * H_);

  // 3. LSTM stack (WMMA gates GEMMs + pointwise)
  gemm32_dual_bf16<<<(4 * H_ / 16 + 7) / 8, 256, 0, stream>>>(
      x_bf, Wih0, 2 * E_, h0_bf, Whh0, H_, bih0, bhh0,
      4 * H_, 4 * H_, 4 * H_, gates, 0);
  lstm_pointwise<<<(B_ * H_) / 256, 256, 0, stream>>>(
      gates, c0, hn, cn, h_bf, h_f32);
  for (int l = 1; l < L_; ++l) {
    const float* Wih = Wih_r + (size_t)(l - 1) * 4 * H_ * H_;
    const float* Whh = Whh_r + (size_t)(l - 1) * 4 * H_ * H_;
    gemm32_dual_bf16<<<(4 * H_ / 16 + 7) / 8, 256, 0, stream>>>(
        h_bf, Wih, H_, h0_bf + (size_t)l * B_ * H_, Whh, H_,
        bih_r + (size_t)(l - 1) * 4 * H_, bhh_r + (size_t)(l - 1) * 4 * H_,
        4 * H_, 4 * H_, 4 * H_, gates, 0);
    lstm_pointwise<<<(B_ * H_) / 256, 256, 0, stream>>>(
        gates, c0 + (size_t)l * B_ * H_, hn + (size_t)l * B_ * H_,
        cn + (size_t)l * B_ * H_, h_bf, h_f32);
  }

  // 4. attention (algebraically collapsed to two streaming passes over enc)
  attn_prep<<<(2 * H_) / 256, 256, 0, stream>>>(Wa, vv, uh, ue);
  qconst_kernel<<<B_ / 8, 256, 0, stream>>>(h_f32, uh, ba, vv, qc);
  energy_kernel<<<(B_ * S_) / 8, 256, 0, stream>>>(enc, ue, qc, energy);
  softmax_kernel<<<B_, 256, 0, stream>>>(energy, attns);
  context_partial<<<dim3(16, B_), 256, 0, stream>>>(enc, attns, part);
  context_reduce<<<(B_ * H_) / 256, 256, 0, stream>>>(part, ctx);

  // 5. hidden = tanh([out|context|key_e] @ Wal^T + bal)   (WMMA, fused tanh)
  build_tmp<<<(B_ * 3072) / 256, 256, 0, stream>>>(h_f32, ctx, x_f32, tmp_bf);
  gemm32_dual_bf16<<<(H_ / 16 + 7) / 8, 256, 0, stream>>>(
      tmp_bf, Wal, 3072, nullptr, nullptr, 0, bal, nullptr,
      H_, H_, H_, hidden_f32, 1);
  cvt_bf16<<<(B_ * H_) / 256, 256, 0, stream>>>(hidden_f32, hidden_bf, B_ * H_);

  // 6. logits = hidden @ Wout^T + bout  (WMMA, N padded to 16, guarded)
  const int ntiles = (V_ + 15) / 16;          // 3142
  const int npad = ntiles * 16;               // 50272
  gemm32_dual_bf16<<<(ntiles + 7) / 8, 256, 0, stream>>>(
      hidden_bf, Wout, H_, nullptr, nullptr, 0, bout, nullptr,
      npad, V_, V_, logits, 0);
}
